// ViTVQVAE_8581344657946
// MI455X (gfx1250) — compile-verified, matrix operand order checked by reading, not run on hardware
//
#include <hip/hip_runtime.h>
#include <stdint.h>

// ---------------------------------------------------------------------------
// ViT-VQVAE forward for MI455X (gfx1250, wave32).
// All GEMMs (incl. the whole deconv decoder as implicit GEMM) use
// v_wmma_f32_16x16x32_bf16 with fp32 accumulation; tile staging uses
// GLOBAL_LOAD_ASYNC_TO_LDS_B128 with double-buffered LDS.
// ---------------------------------------------------------------------------

typedef __bf16 bf16_t;
typedef __attribute__((ext_vector_type(16))) __bf16 v16bf;
typedef __attribute__((ext_vector_type(8)))  float  v8f;

// Model constants
#define IMGSZ 512
#define PATCH 32
#define CINC  3
#define DIM   256
#define NCODE 8192
#define NLAY  6
#define NHEAD 8
#define DHEAD 32
#define FFDIM 1024
#define BATCH 32
#define GRIDP 16
#define NPAT  256            // GRIDP*GRIDP
#define ROWS  8192           // BATCH*NPAT
#define PKCOL 3072           // CINC*PATCH*PATCH

static __device__ __forceinline__ unsigned short f2bf(float f) {
  unsigned int u = __float_as_uint(f);
  u += 0x7FFFu + ((u >> 16) & 1u);   // round-to-nearest-even
  return (unsigned short)(u >> 16);
}

// Async global->LDS copy of 16 bytes (CDNA5 async path, ASYNCcnt tracked).
static __device__ __forceinline__ void async_b128(unsigned lds_off,
                                                  const unsigned short* g) {
  asm volatile("global_load_async_to_lds_b128 %0, %1, off"
               :: "v"(lds_off), "v"((unsigned long long)(size_t)g)
               : "memory");
}
static __device__ __forceinline__ void wait_async0() {
  asm volatile("s_wait_asynccnt 0" ::: "memory");
}

// ------------------------------- convert -----------------------------------
__global__ void f32_to_bf16_k(const float* __restrict__ in,
                              unsigned short* __restrict__ out, long n) {
  long i = (long)blockIdx.x * blockDim.x + threadIdx.x;
  long st = (long)gridDim.x * blockDim.x;
  for (; i < n; i += st) out[i] = f2bf(in[i]);
}

// ------------------------------- im2col ------------------------------------
// x (B,3,512,512) fp32 -> A (8192, 3072) bf16, col = c*1024 + i*32 + j
__global__ void im2col_patch_k(const float* __restrict__ x,
                               unsigned short* __restrict__ A) {
  const long n = (long)ROWS * PKCOL;
  long i = (long)blockIdx.x * blockDim.x + threadIdx.x;
  long st = (long)gridDim.x * blockDim.x;
  for (; i < n; i += st) {
    int r = (int)(i / PKCOL);
    int col = (int)(i - (long)r * PKCOL);
    int b = r >> 8, p = r & 255;
    int gy = p >> 4, gx = p & 15;
    int c = col >> 10, rem = col & 1023;
    int ii = rem >> 5, jj = rem & 31;
    A[i] = f2bf(x[(((size_t)b * CINC + c) * IMGSZ + gy * PATCH + ii) * IMGSZ +
                  gx * PATCH + jj]);
  }
}

// ------------------------------- GEMM (WMMA) -------------------------------
// C(MxN,f32) = A(MxK,bf16 row-major) * B(NxK,bf16 row-major)^T + bias
// act: 0=none, 1=exact GELU.  Requires M%64==0, N%64==0, K%32==0.
// Double-buffered LDS tiles filled by async global->LDS copies.
__global__ __launch_bounds__(128)
void gemm_bf16_k(const unsigned short* __restrict__ A,
                 const unsigned short* __restrict__ Bw,
                 const float* __restrict__ bias,
                 float* __restrict__ C,
                 int M, int N, int K, int act) {
  __shared__ uint4 As[2][64 * 4];   // 64 rows x 32 bf16 per buffer
  __shared__ uint4 Bs[2][64 * 4];

  const int tid  = threadIdx.x;
  const int lane = tid & 31;
  const int wv   = tid >> 5;       // wave 0..3
  const int n0   = blockIdx.x * 64;
  const int m0   = blockIdx.y * 64;
  const int m    = lane & 15;
  const int hf   = lane >> 4;

  v8f acc[4];
#pragma unroll
  for (int t = 0; t < 4; ++t)
    acc[t] = (v8f){0.f, 0.f, 0.f, 0.f, 0.f, 0.f, 0.f, 0.f};

  const int lrow = tid >> 1;       // 0..63
  const int lcb  = tid & 1;        // 16-half chunk select

  const unsigned short* garow = A  + (size_t)(m0 + lrow) * K + lcb * 16;
  const unsigned short* gbrow = Bw + (size_t)(n0 + lrow) * K + lcb * 16;

#define ISSUE_TILE(buf, k0)                                                   \
  do {                                                                        \
    unsigned la = (unsigned)(size_t)&As[buf][lrow * 4 + lcb * 2];             \
    unsigned lb = (unsigned)(size_t)&Bs[buf][lrow * 4 + lcb * 2];             \
    async_b128(la,      garow + (k0));                                        \
    async_b128(la + 16, garow + (k0) + 8);                                    \
    async_b128(lb,      gbrow + (k0));                                        \
    async_b128(lb + 16, gbrow + (k0) + 8);                                    \
  } while (0)

  ISSUE_TILE(0, 0);
  wait_async0();
  __syncthreads();

  int buf = 0;
  for (int k0 = 0; k0 < K; k0 += 32) {
    if (k0 + 32 < K) ISSUE_TILE(buf ^ 1, k0 + 32);

    union Fr { uint4 u[2]; v16bf v; };
    Fr af;
    af.u[0] = As[buf][(wv * 16 + m) * 4 + hf];
    af.u[1] = As[buf][(wv * 16 + m) * 4 + 2 + hf];
#pragma unroll
    for (int nt = 0; nt < 4; ++nt) {
      Fr bf;
      bf.u[0] = Bs[buf][(nt * 16 + m) * 4 + hf];
      bf.u[1] = Bs[buf][(nt * 16 + m) * 4 + 2 + hf];
      acc[nt] = __builtin_amdgcn_wmma_f32_16x16x32_bf16(
          false, af.v, false, bf.v, (short)0, acc[nt], false, false);
    }
    wait_async0();
    __syncthreads();
    buf ^= 1;
  }
#undef ISSUE_TILE

  const int crow = m0 + wv * 16 + 8 * hf;
#pragma unroll
  for (int nt = 0; nt < 4; ++nt) {
    const int n = n0 + nt * 16 + m;
    const float bv = bias ? bias[n] : 0.f;
#pragma unroll
    for (int g = 0; g < 8; ++g) {
      float v = acc[nt][g] + bv;
      if (act == 1) v = 0.5f * v * (1.0f + erff(v * 0.70710678118654752f));
      C[(size_t)(crow + g) * N + n] = v;
    }
  }
}

// ----------------------- deconv as implicit GEMM (WMMA) --------------------
// ConvTranspose2d(stride2,k4,p1) parity class (ry,rx):
//   out[b, co, 2q+ry, 2p+rx] = sum_{ci, dy, dx in {0,1}}
//       in[b, q+ry-1+dy, p+rx-1+dx, ci] * Wp[co, (dy*2+dx)*Cin + ci]
// In is NHWC bf16; Wp is the parity-repacked (CoutPad x 4Cin) bf16 weight.
// mode 2: relu -> bf16 NHWC out (Cout == CoutPad)
// mode 3: sigmoid -> fp32 NCHW scatter, first coutReal channels only.
__global__ __launch_bounds__(128)
void deconv_gemm_k(const unsigned short* __restrict__ In,
                   const unsigned short* __restrict__ Wp,
                   const float* __restrict__ bias, int coutReal,
                   void* __restrict__ Out,
                   int Hi, int Wi, int Cin, int CoutPad,
                   int ry, int rx, int mode) {
  const int K = 4 * Cin;
  const int HW = Hi * Wi;
  __shared__ uint4 As[2][64 * 4];
  __shared__ uint4 Bs[2][64 * 4];

  const int tid  = threadIdx.x;
  const int lane = tid & 31;
  const int wv   = tid >> 5;
  const int n0   = blockIdx.x * 64;
  const int m0   = blockIdx.y * 64;
  const int m    = lane & 15;
  const int hf   = lane >> 4;

  v8f acc[4];
#pragma unroll
  for (int t = 0; t < 4; ++t)
    acc[t] = (v8f){0.f, 0.f, 0.f, 0.f, 0.f, 0.f, 0.f, 0.f};

  const int lrow = tid >> 1;
  const int lcb  = tid & 1;

  // pixel this thread gathers for the A tile
  const int arow = m0 + lrow;
  const int ab   = arow / HW;
  const int arem = arow - ab * HW;
  const int aq   = arem / Wi;
  const int ap   = arem - aq * Wi;
  const unsigned short* gbrow = Wp + (size_t)(n0 + lrow) * K + lcb * 16;

#define ISSUE_DC(buf, k0)                                                     \
  do {                                                                        \
    const int k   = (k0) + lcb * 16;                                          \
    const int win = k / Cin;                                                  \
    const int ci0 = k - win * Cin;                                            \
    const int dy  = win >> 1, dx = win & 1;                                   \
    const int iy  = aq + ry - 1 + dy;                                         \
    const int ix  = ap + rx - 1 + dx;                                         \
    unsigned la = (unsigned)(size_t)&As[buf][lrow * 4 + lcb * 2];             \
    unsigned lb = (unsigned)(size_t)&Bs[buf][lrow * 4 + lcb * 2];             \
    if (iy >= 0 && iy < Hi && ix >= 0 && ix < Wi) {                           \
      const unsigned short* g =                                               \
          In + (((size_t)ab * Hi + iy) * Wi + ix) * Cin + ci0;                \
      async_b128(la, g);                                                      \
      async_b128(la + 16, g + 8);                                             \
    } else {                                                                  \
      uint4 z = {0u, 0u, 0u, 0u};                                             \
      As[buf][lrow * 4 + lcb * 2]     = z;                                    \
      As[buf][lrow * 4 + lcb * 2 + 1] = z;                                    \
    }                                                                         \
    async_b128(lb, gbrow + (k0));                                             \
    async_b128(lb + 16, gbrow + (k0) + 8);                                    \
  } while (0)

  ISSUE_DC(0, 0);
  wait_async0();
  __syncthreads();

  int buf = 0;
  for (int k0 = 0; k0 < K; k0 += 32) {
    if (k0 + 32 < K) ISSUE_DC(buf ^ 1, k0 + 32);

    union Fr { uint4 u[2]; v16bf v; };
    Fr af;
    af.u[0] = As[buf][(wv * 16 + m) * 4 + hf];
    af.u[1] = As[buf][(wv * 16 + m) * 4 + 2 + hf];
#pragma unroll
    for (int nt = 0; nt < 4; ++nt) {
      Fr bf;
      bf.u[0] = Bs[buf][(nt * 16 + m) * 4 + hf];
      bf.u[1] = Bs[buf][(nt * 16 + m) * 4 + 2 + hf];
      acc[nt] = __builtin_amdgcn_wmma_f32_16x16x32_bf16(
          false, af.v, false, bf.v, (short)0, acc[nt], false, false);
    }
    wait_async0();
    __syncthreads();
    buf ^= 1;
  }
#undef ISSUE_DC

  // decode output pixels for this lane's 8 rows
  int rb[8], oy[8], ox[8];
#pragma unroll
  for (int g = 0; g < 8; ++g) {
    const int row = m0 + wv * 16 + 8 * hf + g;
    const int b   = row / HW;
    const int rem = row - b * HW;
    const int q   = rem / Wi;
    const int p   = rem - q * Wi;
    rb[g] = b; oy[g] = 2 * q + ry; ox[g] = 2 * p + rx;
  }
  const int Ho = 2 * Hi, Wo = 2 * Wi;
#pragma unroll
  for (int nt = 0; nt < 4; ++nt) {
    const int n = n0 + nt * 16 + m;
    const float bv = (n < coutReal) ? bias[n] : 0.f;
#pragma unroll
    for (int g = 0; g < 8; ++g) {
      float v = acc[nt][g] + bv;
      if (mode == 2) {
        v = fmaxf(v, 0.f);
        ((unsigned short*)Out)[(((size_t)rb[g] * Ho + oy[g]) * Wo + ox[g]) *
                                   CoutPad + n] = f2bf(v);
      } else if (n < coutReal) {
        v = 1.0f / (1.0f + __expf(-v));
        ((float*)Out)[(((size_t)rb[g] * coutReal + n) * Ho + oy[g]) * Wo +
                      ox[g]] = v;
      }
    }
  }
}

// ----------------- deconv weight repack: parity-major bf16 -----------------
// wp[(parity*CoutPad + co)*4Cin + win*Cin + ci] = w[ci, co, ky, kx]
// dy=0 -> ky=3-ry (iy=q+ry-1); dy=1 -> ky=1-ry (iy=q+ry); same for x.
__global__ void repack_deconv_w_k(const float* __restrict__ w,
                                  unsigned short* __restrict__ wp,
                                  int Cin, int Cout, int CoutPad) {
  const long total = 4L * CoutPad * 4 * Cin;
  long i = (long)blockIdx.x * blockDim.x + threadIdx.x;
  long st = (long)gridDim.x * blockDim.x;
  for (; i < total; i += st) {
    const int par = (int)(i / ((long)CoutPad * 4 * Cin));
    const long rem = i - (long)par * CoutPad * 4 * Cin;
    const int co = (int)(rem / (4 * Cin));
    const int k  = (int)(rem - (long)co * 4 * Cin);
    const int win = k / Cin, ci = k - win * Cin;
    const int ry = par >> 1, rx = par & 1;
    const int dy = win >> 1, dx = win & 1;
    const int ky = (dy == 0) ? (3 - ry) : (1 - ry);
    const int kx = (dx == 0) ? (3 - rx) : (1 - rx);
    float v = 0.f;
    if (co < Cout) v = w[(((size_t)ci * Cout + co) * 4 + ky) * 4 + kx];
    wp[i] = f2bf(v);
  }
}

// ------------------------------- attention ---------------------------------
__global__ __launch_bounds__(256)
void attention_k(const float* __restrict__ qkv, float* __restrict__ out) {
  __shared__ float kb[8][32][33];
  __shared__ float vb[8][32][33];
  const int lane = threadIdx.x & 31;           // s (image index)
  const int wv   = threadIdx.x >> 5;
  const int pair = blockIdx.x * 8 + wv;        // bt*8 + head
  const int head = pair & 7;
  const int bt   = pair >> 3;
  const size_t rb = ((size_t)lane * NPAT + bt) * (3 * DIM) + head * DHEAD;

  float q[DHEAD], sc[32];
#pragma unroll
  for (int d = 0; d < DHEAD; ++d) {
    q[d] = qkv[rb + d];
    kb[wv][lane][d] = qkv[rb + DIM + d];
    vb[wv][lane][d] = qkv[rb + 2 * DIM + d];
  }
  __syncthreads();

  float mx = -3.4e38f;
#pragma unroll 4
  for (int t = 0; t < 32; ++t) {
    float s = 0.f;
#pragma unroll
    for (int d = 0; d < DHEAD; ++d) s += q[d] * kb[wv][t][d];
    s *= 0.17677669529663687f;   // 1/sqrt(32)
    sc[t] = s;
    mx = fmaxf(mx, s);
  }
  float sum = 0.f;
#pragma unroll 4
  for (int t = 0; t < 32; ++t) { sc[t] = __expf(sc[t] - mx); sum += sc[t]; }
  const float inv = 1.0f / sum;
#pragma unroll 2
  for (int d = 0; d < DHEAD; ++d) {
    float o = 0.f;
#pragma unroll
    for (int t = 0; t < 32; ++t) o += sc[t] * vb[wv][t][d];
    out[((size_t)lane * NPAT + bt) * DIM + head * DHEAD + d] = o * inv;
  }
}

// --------------------------- residual + layernorm --------------------------
__global__ __launch_bounds__(256)
void resid_ln_k(float* __restrict__ h, const float* __restrict__ add,
                const float* __restrict__ w, const float* __restrict__ b) {
  const int lane = threadIdx.x & 31;
  const int row  = blockIdx.x * 8 + (threadIdx.x >> 5);
  const size_t base = (size_t)row * DIM;
  float x[8];
  float s = 0.f, ss = 0.f;
#pragma unroll
  for (int j = 0; j < 8; ++j) {
    const int d = j * 32 + lane;
    float v = h[base + d];
    if (add) v += add[base + d];
    x[j] = v; s += v; ss += v * v;
  }
#pragma unroll
  for (int off = 1; off < 32; off <<= 1) {
    s  += __shfl_xor(s, off, 32);
    ss += __shfl_xor(ss, off, 32);
  }
  const float mean = s * (1.0f / DIM);
  const float var  = ss * (1.0f / DIM) - mean * mean;
  const float inv  = rsqrtf(var + 1e-5f);
#pragma unroll
  for (int j = 0; j < 8; ++j) {
    const int d = j * 32 + lane;
    h[base + d] = (x[j] - mean) * inv * w[d] + b[d];
  }
}

// ------------------------------- code norms --------------------------------
__global__ void code_norm_k(const float* __restrict__ cb, float* __restrict__ cn) {
  const int k = blockIdx.x * blockDim.x + threadIdx.x;
  if (k < NCODE) {
    float s = 0.f;
    for (int d = 0; d < DIM; ++d) { float v = cb[(size_t)k * DIM + d]; s += v * v; }
    cn[k] = s;
  }
}

// ------------------------------ fused VQ argmin ----------------------------
// dist_k = ||c_k||^2 - 2 h.c_k (h-norm irrelevant for argmin).
__global__ __launch_bounds__(32)
void vq_argmin_k(const unsigned short* __restrict__ Hb,
                 const unsigned short* __restrict__ Cb,
                 const float* __restrict__ cnorm,
                 int* __restrict__ idx) {
  const int lane = threadIdx.x & 31;
  const int m  = lane & 15;
  const int hf = lane >> 4;
  const int m0 = blockIdx.x * 16;

  union Fr { uint4 u[2]; v16bf v; };
  Fr af[8];
#pragma unroll
  for (int kk = 0; kk < 8; ++kk) {
    const uint4* p =
        reinterpret_cast<const uint4*>(Hb + (size_t)(m0 + m) * DIM + kk * 32) + hf;
    af[kk].u[0] = p[0];
    af[kk].u[1] = p[2];
  }

  float best[8]; int bidx[8];
#pragma unroll
  for (int g = 0; g < 8; ++g) { best[g] = 3.4e38f; bidx[g] = 0; }

  for (int c0 = 0; c0 < NCODE; c0 += 16) {
    v8f acc = (v8f){0.f, 0.f, 0.f, 0.f, 0.f, 0.f, 0.f, 0.f};
#pragma unroll
    for (int kk = 0; kk < 8; ++kk) {
      Fr bf;
      const uint4* p =
          reinterpret_cast<const uint4*>(Cb + (size_t)(c0 + m) * DIM + kk * 32) + hf;
      bf.u[0] = p[0];
      bf.u[1] = p[2];
      acc = __builtin_amdgcn_wmma_f32_16x16x32_bf16(
          false, af[kk].v, false, bf.v, (short)0, acc, false, false);
    }
    const float cn = cnorm[c0 + m];
#pragma unroll
    for (int g = 0; g < 8; ++g) {
      float d = cn - 2.0f * acc[g];
      int   ci = c0 + m;
#pragma unroll
      for (int off = 1; off < 16; off <<= 1) {
        float od = __shfl_xor(d, off, 32);
        int   oi = __shfl_xor(ci, off, 32);
        if (od < d || (od == d && oi < ci)) { d = od; ci = oi; }
      }
      if (d < best[g] || (d == best[g] && ci < bidx[g])) { best[g] = d; bidx[g] = ci; }
    }
  }
  if (m == 0) {
#pragma unroll
    for (int g = 0; g < 8; ++g) idx[m0 + g + 8 * hf] = bidx[g];
  }
}

// ----------------- gather q (bf16 NHWC for decoder) + vq loss --------------
// Writes one deterministic partial sum per block (no atomics).
__global__ __launch_bounds__(256)
void gather_q_k(const int* __restrict__ idx, const float* __restrict__ cb,
                const float* __restrict__ h, unsigned short* __restrict__ qn,
                float* __restrict__ partial) {
  __shared__ float red[256];
  const long t = (long)blockIdx.x * 256 + threadIdx.x;  // ROWS*DIM threads
  const int r = (int)(t >> 8);
  const int d = (int)(t & 255);
  const int b = r >> 8, p = r & 255;
  const int y = p >> 4, xx = p & 15;
  const float qv = cb[(size_t)idx[r] * DIM + d];
  qn[(((size_t)b * GRIDP + y) * GRIDP + xx) * DIM + d] = f2bf(qv);
  const float df = qv - h[t];
  red[threadIdx.x] = df * df;
  __syncthreads();
  for (int s = 128; s > 0; s >>= 1) {
    if (threadIdx.x < s) red[threadIdx.x] += red[threadIdx.x + s];
    __syncthreads();
  }
  if (threadIdx.x == 0) partial[blockIdx.x] = red[0];
}

// Deterministic final reduction of ROWS partials (single block, fixed order).
__global__ __launch_bounds__(256)
void finalize_loss_k(const float* __restrict__ partial,
                     float* __restrict__ out) {
  __shared__ float red[256];
  float s = 0.f;
  for (int i = threadIdx.x; i < ROWS; i += 256) s += partial[i];
  red[threadIdx.x] = s;
  __syncthreads();
  for (int st = 128; st > 0; st >>= 1) {
    if (threadIdx.x < st) red[threadIdx.x] += red[threadIdx.x + st];
    __syncthreads();
  }
  if (threadIdx.x == 0)
    out[0] = 1.25f * red[0] * (1.0f / ((float)ROWS * (float)DIM));
}

// ---------------------------------------------------------------------------
extern "C" void kernel_launch(void* const* d_in, const int* in_sizes, int n_in,
                              void* d_out, int out_size, void* d_ws, size_t ws_size,
                              hipStream_t stream) {
  const float* x       = (const float*)d_in[0];
  const float* patch_w = (const float*)d_in[1];
  const float* patch_b = (const float*)d_in[2];
  const float* qkv_w   = (const float*)d_in[3];
  const float* qkv_b   = (const float*)d_in[4];
  const float* out_w   = (const float*)d_in[5];
  const float* out_b   = (const float*)d_in[6];
  const float* ln1_w   = (const float*)d_in[7];
  const float* ln1_b   = (const float*)d_in[8];
  const float* ln2_w   = (const float*)d_in[9];
  const float* ln2_b   = (const float*)d_in[10];
  const float* ff1_w   = (const float*)d_in[11];
  const float* ff1_b   = (const float*)d_in[12];
  const float* ff2_w   = (const float*)d_in[13];
  const float* ff2_b   = (const float*)d_in[14];
  const float* lnf_w   = (const float*)d_in[15];
  const float* lnf_b   = (const float*)d_in[16];
  const float* codebk  = (const float*)d_in[17];
  const float* dw0 = (const float*)d_in[18]; const float* db0 = (const float*)d_in[19];
  const float* dw1 = (const float*)d_in[20]; const float* db1 = (const float*)d_in[21];
  const float* dw2 = (const float*)d_in[22]; const float* db2 = (const float*)d_in[23];
  const float* dw3 = (const float*)d_in[24]; const float* db3 = (const float*)d_in[25];

  char* ws = (char*)d_ws;
  // ---- workspace layout (~256 MB) ----
  unsigned short* Apatch = (unsigned short*)(ws + 0);              // 48 MB
  unsigned short* wPatch = (unsigned short*)(ws + 50331648);       // 16 MB weights
  unsigned short* wQkv  = wPatch + (size_t)DIM * PKCOL;
  unsigned short* wOut  = wQkv  + (size_t)NLAY * 3 * DIM * DIM;
  unsigned short* wFf1  = wOut  + (size_t)NLAY * DIM * DIM;
  unsigned short* wFf2  = wFf1  + (size_t)NLAY * FFDIM * DIM;
  unsigned short* wCode = wFf2  + (size_t)NLAY * DIM * FFDIM;
  float* cnorm          = (float*)(wCode + (size_t)NCODE * DIM);
  float* h              = (float*)(ws + 67108864);                 // 8 MB
  char*  S              = ws + 75497472;                           // 64 MB scratch
  unsigned short* s_hb  = (unsigned short*)S;                      // 16 MB staging
  float* s_f32a         = (float*)(S + 16777216);                  // 32 MB
  float* s_f32b         = (float*)(S + 50331648);                  // 8 MB
  int*   s_idx          = (int*)(S + 58720256);                    // 32 KB
  float* s_part         = (float*)(S + 58720256 + 32768);          // 32 KB partials
  char*  DW             = ws + 142606336;                          // decoder region
  unsigned short* wp0   = (unsigned short*)(DW);                   // 2 MB
  unsigned short* wp1   = (unsigned short*)(DW + 2097152);         // 1 MB
  unsigned short* wp2   = (unsigned short*)(DW + 3145728);         // 256 KB
  unsigned short* wp3   = (unsigned short*)(DW + 3407872);         // 128 KB
  unsigned short* qn    = (unsigned short*)(DW + 4194304);         // 4 MB  (B,16,16,256)
  unsigned short* dn0   = (unsigned short*)(DW + 8388608);         // 16 MB (B,32,32,256)
  unsigned short* dn1   = (unsigned short*)(DW + 25165824);        // 32 MB (B,64,64,128)
  unsigned short* dn2   = (unsigned short*)(DW + 58720256);        // 64 MB (B,128,128,64)
  float* xhat           = (float*)d_out;                           // (B,3,256,256)
  float* loss_out       = xhat + (size_t)BATCH * CINC * 256 * 256;

  // ---- 1) weight conversion / repack ----
  f32_to_bf16_k<<<1024, 256, 0, stream>>>(patch_w, wPatch, (long)DIM * PKCOL);
  f32_to_bf16_k<<<1024, 256, 0, stream>>>(qkv_w, wQkv, (long)NLAY * 3 * DIM * DIM);
  f32_to_bf16_k<<<1024, 256, 0, stream>>>(out_w, wOut, (long)NLAY * DIM * DIM);
  f32_to_bf16_k<<<1024, 256, 0, stream>>>(ff1_w, wFf1, (long)NLAY * FFDIM * DIM);
  f32_to_bf16_k<<<1024, 256, 0, stream>>>(ff2_w, wFf2, (long)NLAY * DIM * FFDIM);
  f32_to_bf16_k<<<1024, 256, 0, stream>>>(codebk, wCode, (long)NCODE * DIM);
  code_norm_k<<<NCODE / 256, 256, 0, stream>>>(codebk, cnorm);
  repack_deconv_w_k<<<1024, 256, 0, stream>>>(dw0, wp0, 256, 256, 256);
  repack_deconv_w_k<<<1024, 256, 0, stream>>>(dw1, wp1, 256, 128, 128);
  repack_deconv_w_k<<<512,  256, 0, stream>>>(dw2, wp2, 128, 64, 64);
  repack_deconv_w_k<<<256,  256, 0, stream>>>(dw3, wp3, 64, 3, 64);

  // ---- 2) patch embedding ----
  im2col_patch_k<<<49152, 256, 0, stream>>>(x, Apatch);
  gemm_bf16_k<<<dim3(DIM / 64, ROWS / 64), 128, 0, stream>>>(
      Apatch, wPatch, patch_b, h, ROWS, DIM, PKCOL, 0);

  // ---- 3) transformer layers ----
  for (int i = 0; i < NLAY; ++i) {
    f32_to_bf16_k<<<2048, 256, 0, stream>>>(h, s_hb, (long)ROWS * DIM);
    gemm_bf16_k<<<dim3((3 * DIM) / 64, ROWS / 64), 128, 0, stream>>>(
        s_hb, wQkv + (size_t)i * 3 * DIM * DIM, qkv_b + (size_t)i * 3 * DIM,
        s_f32a, ROWS, 3 * DIM, DIM, 0);
    attention_k<<<(NPAT * NHEAD) / 8, 256, 0, stream>>>(s_f32a, s_f32b);
    f32_to_bf16_k<<<2048, 256, 0, stream>>>(s_f32b, s_hb, (long)ROWS * DIM);
    gemm_bf16_k<<<dim3(DIM / 64, ROWS / 64), 128, 0, stream>>>(
        s_hb, wOut + (size_t)i * DIM * DIM, out_b + (size_t)i * DIM,
        s_f32a, ROWS, DIM, DIM, 0);
    resid_ln_k<<<ROWS / 8, 256, 0, stream>>>(h, s_f32a,
                                             ln1_w + (size_t)i * DIM,
                                             ln1_b + (size_t)i * DIM);
    f32_to_bf16_k<<<2048, 256, 0, stream>>>(h, s_hb, (long)ROWS * DIM);
    gemm_bf16_k<<<dim3(FFDIM / 64, ROWS / 64), 128, 0, stream>>>(
        s_hb, wFf1 + (size_t)i * FFDIM * DIM, ff1_b + (size_t)i * FFDIM,
        s_f32a, ROWS, FFDIM, DIM, 1);
    f32_to_bf16_k<<<4096, 256, 0, stream>>>(s_f32a, s_hb, (long)ROWS * FFDIM);
    gemm_bf16_k<<<dim3(DIM / 64, ROWS / 64), 128, 0, stream>>>(
        s_hb, wFf2 + (size_t)i * DIM * FFDIM, ff2_b + (size_t)i * DIM,
        s_f32b, ROWS, DIM, FFDIM, 0);
    resid_ln_k<<<ROWS / 8, 256, 0, stream>>>(h, s_f32b,
                                             ln2_w + (size_t)i * DIM,
                                             ln2_b + (size_t)i * DIM);
  }

  // ---- 4) final LN + fused VQ argmin ----
  resid_ln_k<<<ROWS / 8, 256, 0, stream>>>(h, nullptr, lnf_w, lnf_b);
  f32_to_bf16_k<<<2048, 256, 0, stream>>>(h, s_hb, (long)ROWS * DIM);
  vq_argmin_k<<<ROWS / 16, 32, 0, stream>>>(s_hb, wCode, cnorm, s_idx);

  // ---- 5) gather q (bf16 NHWC) + deterministic vq loss ----
  gather_q_k<<<ROWS, 256, 0, stream>>>(s_idx, codebk, h, qn, s_part);
  finalize_loss_k<<<1, 256, 0, stream>>>(s_part, loss_out);

  // ---- 6) decoder: 4 deconv layers x 4 parity-class implicit GEMMs ----
  for (int par = 0; par < 4; ++par) {
    const int ry = par >> 1, rx = par & 1;
    deconv_gemm_k<<<dim3(4, 128), 128, 0, stream>>>(
        qn, wp0 + (size_t)par * 256 * 1024, db0, 256, dn0,
        16, 16, 256, 256, ry, rx, 2);
  }
  for (int par = 0; par < 4; ++par) {
    const int ry = par >> 1, rx = par & 1;
    deconv_gemm_k<<<dim3(2, 512), 128, 0, stream>>>(
        dn0, wp1 + (size_t)par * 128 * 1024, db1, 128, dn1,
        32, 32, 256, 128, ry, rx, 2);
  }
  for (int par = 0; par < 4; ++par) {
    const int ry = par >> 1, rx = par & 1;
    deconv_gemm_k<<<dim3(1, 2048), 128, 0, stream>>>(
        dn1, wp2 + (size_t)par * 64 * 512, db2, 64, dn2,
        64, 64, 128, 64, ry, rx, 2);
  }
  for (int par = 0; par < 4; ++par) {
    const int ry = par >> 1, rx = par & 1;
    deconv_gemm_k<<<dim3(1, 8192), 128, 0, stream>>>(
        dn2, wp3 + (size_t)par * 64 * 256, db3, 3, xhat,
        128, 128, 64, 64, ry, rx, 3);
  }
}
